// DecoderLSTM_74105365725201
// MI455X (gfx1250) — matrix operand order repaired and checked by reading, not compiled, not run on hardware
//
#include <hip/hip_runtime.h>
#include <stdint.h>

// ---- problem constants ----
#define B_   256
#define H_   1024
#define L_   2
#define T_   96
#define G4H  4096        // 4*H
#define K2H  2048        // 2*H  (concat [x|h] K dimension)
#define NKT  (K2H / 32)  // 64 K-iterations

typedef __attribute__((ext_vector_type(16))) __bf16 bf16x16;
typedef __attribute__((ext_vector_type(8)))  float  f32x8;
typedef __attribute__((ext_vector_type(4)))  int    v4i;

union FragU { bf16x16 bf; uint4 q[2]; };

#if defined(__AMDGCN__) && __has_builtin(__builtin_amdgcn_global_load_async_to_lds_b128)
#define HAVE_ASYNC_LDS 1
#else
#define HAVE_ASYNC_LDS 0
#endif

__device__ __forceinline__ unsigned short f32_to_bf16(float f) {
    unsigned int u = __float_as_uint(f);
    u += 0x7FFFu + ((u >> 16) & 1u);          // round to nearest even
    return (unsigned short)(u >> 16);
}
__device__ __forceinline__ float sigmf(float x) { return 1.0f / (1.0f + __expf(-x)); }

// ---------------------------------------------------------------------------
// One-time: pack [Wih | Whh] -> bf16 Wcat[l] of shape (4H, 2H), row-major.
// ---------------------------------------------------------------------------
__global__ void pack_weights(const float* __restrict__ Wih,
                             const float* __restrict__ Whh,
                             unsigned short* __restrict__ Wcat) {
    const int total = L_ * G4H * K2H;
    for (int idx = blockIdx.x * blockDim.x + threadIdx.x; idx < total;
         idx += gridDim.x * blockDim.x) {
        int l   = idx / (G4H * K2H);
        int rem = idx - l * (G4H * K2H);
        int n   = rem / K2H;
        int k   = rem - n * K2H;
        float v = (k < H_) ? Wih[((size_t)l * G4H + n) * H_ + k]
                           : Whh[((size_t)l * G4H + n) * H_ + (k - H_)];
        Wcat[idx] = f32_to_bf16(v);
    }
}

// ---------------------------------------------------------------------------
// One-time: copy initial states into d_out state regions, build bias sums,
// seed the bf16 [x|h] activation buffers.
// ---------------------------------------------------------------------------
__global__ void init_state(const float* __restrict__ x0,
                           const float* __restrict__ h0,
                           const float* __restrict__ c0,
                           const float* __restrict__ bih,
                           const float* __restrict__ bhh,
                           float* __restrict__ hst, float* __restrict__ cst,
                           float* __restrict__ bsum,
                           unsigned short* __restrict__ xh0,
                           unsigned short* __restrict__ xh1) {
    int idx = blockIdx.x * blockDim.x + threadIdx.x;
    if (idx < L_ * B_ * H_) { hst[idx] = h0[idx]; cst[idx] = c0[idx]; }
    if (idx < B_ * H_) {
        int b = idx >> 10, k = idx & (H_ - 1);
        xh0[b * K2H + k]      = f32_to_bf16(x0[idx]);               // x part (t=0)
        xh0[b * K2H + H_ + k] = f32_to_bf16(h0[idx]);               // layer0 h state
        xh1[b * K2H + H_ + k] = f32_to_bf16(h0[B_ * H_ + idx]);     // layer1 h state
        xh1[b * K2H + k]      = 0;                                  // overwritten each step
    }
    if (idx < L_ * G4H) bsum[idx] = bih[idx] + bhh[idx];
}

// ---------------------------------------------------------------------------
// Fused layer step: gates = xh(256x2048 bf16) @ Wcat_l^T (4096x2048 bf16),
// then LSTM cell update for a (128 batch x 16 hidden) tile per block.
// grid = (H/16, B/128), block = 256 threads (8 wave32).
// Double-buffered LDS, one barrier per K-iteration; next tile staged via
// GLOBAL_LOAD_ASYNC_TO_LDS (ASYNCcnt) when available.
// ---------------------------------------------------------------------------
__global__ __launch_bounds__(256)
void lstm_layer_step(const unsigned short* __restrict__ xh,    // (B, 2H) bf16 [x|h]
                     const unsigned short* __restrict__ Wl,    // (4H, 2H) bf16
                     const float* __restrict__ bsum,           // (4H,)  bih+bhh
                     float* __restrict__ hst,                  // (B,H) f32 (layer slice)
                     float* __restrict__ cst,                  // (B,H) f32
                     unsigned short* __restrict__ hdst1,       // bf16, row stride 2H
                     unsigned short* __restrict__ hdst2)       // bf16, row stride 2H
{
    __shared__ uint4 Alds[2][512];   // 2 x (128 rows x 32 bf16)
    __shared__ uint4 Wlds[2][256];   // 2 x (4 gates x 16 rows x 32 bf16)

    const int tid    = threadIdx.x;
    const int lane   = tid & 31;
    const int wave   = tid >> 5;
    const int laneHi = lane >> 4;      // 0/1
    const int ln     = lane & 15;
    const int n0     = blockIdx.x * 16;    // hidden tile base
    const int m0     = blockIdx.y * 128;   // batch tile base

    const uint4* gx = (const uint4*)xh;    // row = 256 uint4 (2048 bf16)
    const uint4* gw = (const uint4*)Wl;

    // per-thread cooperative-load source indices (A: 2 chunks, W: 1 chunk)
    const int e0 = tid, e1 = tid + 256;
    const int aidx0 = (m0 + (e0 >> 2)) * 256 + (e0 & 3);
    const int aidx1 = (m0 + (e1 >> 2)) * 256 + (e1 & 3);
    const int widx  = (((tid >> 6) * H_) + n0 + ((tid >> 2) & 15)) * 256 + (tid & 3);

    f32x8 acc0 = {}; f32x8 acc1 = {}; f32x8 acc2 = {}; f32x8 acc3 = {};

#if HAVE_ASYNC_LDS
    typedef __attribute__((address_space(1))) v4i g_v4i;
    typedef __attribute__((address_space(3))) v4i l_v4i;
    g_v4i* gxa = (g_v4i*)gx;
    g_v4i* gwa = (g_v4i*)gw;
    l_v4i* lA  = (l_v4i*)&Alds[0][0];     // 2*512 entries
    l_v4i* lW  = (l_v4i*)&Wlds[0][0];     // 2*256 entries
    #define ISSUE_TILE(kt, buf)                                                       \
        do {                                                                          \
            __builtin_amdgcn_global_load_async_to_lds_b128(                           \
                gxa + aidx0 + (kt) * 4, lA + (buf) * 512 + e0, 0, 0);                 \
            __builtin_amdgcn_global_load_async_to_lds_b128(                           \
                gxa + aidx1 + (kt) * 4, lA + (buf) * 512 + e1, 0, 0);                 \
            __builtin_amdgcn_global_load_async_to_lds_b128(                           \
                gwa + widx + (kt) * 4, lW + (buf) * 256 + tid, 0, 0);                 \
        } while (0)
    #define WAIT_TILE()                                                               \
        do {                                                                          \
            __builtin_amdgcn_s_wait_asynccnt(0);                                      \
        } while (0)
#else
    #define ISSUE_TILE(kt, buf)                                                       \
        do {                                                                          \
            uint4 a0 = gx[aidx0 + (kt) * 4];                                          \
            uint4 a1 = gx[aidx1 + (kt) * 4];                                          \
            uint4 w0 = gw[widx + (kt) * 4];                                           \
            Alds[buf][e0] = a0;                                                       \
            Alds[buf][e1] = a1;                                                       \
            Wlds[buf][tid] = w0;                                                      \
        } while (0)
    #define WAIT_TILE() do { } while (0)
#endif

    ISSUE_TILE(0, 0);

    const int arow  = wave * 16 + ln;
    const int wbase = ln * 4 + laneHi * 2;

    for (int kt = 0; kt < NKT; ++kt) {
        const int cur = kt & 1;
        WAIT_TILE();            // my async stores to LDS have landed
        __syncthreads();        // everyone's have landed; prev compute done
        if (kt + 1 < NKT) ISSUE_TILE(kt + 1, cur ^ 1);   // overlaps WMMAs below

        // A fragment: 16 contiguous K per lane-half
        FragU a;
        a.q[0] = Alds[cur][arow * 4 + laneHi * 2];
        a.q[1] = Alds[cur][arow * 4 + laneHi * 2 + 1];

        // B fragments: lane = column, 16 contiguous K per lane-half
        FragU b;
        b.q[0] = Wlds[cur][0 * 64 + wbase]; b.q[1] = Wlds[cur][0 * 64 + wbase + 1];
        acc0 = __builtin_amdgcn_wmma_f32_16x16x32_bf16(false, a.bf, false, b.bf,
                                                       (short)0, acc0, false, false);
        b.q[0] = Wlds[cur][1 * 64 + wbase]; b.q[1] = Wlds[cur][1 * 64 + wbase + 1];
        acc1 = __builtin_amdgcn_wmma_f32_16x16x32_bf16(false, a.bf, false, b.bf,
                                                       (short)0, acc1, false, false);
        b.q[0] = Wlds[cur][2 * 64 + wbase]; b.q[1] = Wlds[cur][2 * 64 + wbase + 1];
        acc2 = __builtin_amdgcn_wmma_f32_16x16x32_bf16(false, a.bf, false, b.bf,
                                                       (short)0, acc2, false, false);
        b.q[0] = Wlds[cur][3 * 64 + wbase]; b.q[1] = Wlds[cur][3 * 64 + wbase + 1];
        acc3 = __builtin_amdgcn_wmma_f32_16x16x32_bf16(false, a.bf, false, b.bf,
                                                       (short)0, acc3, false, false);
    }

    // ---- LSTM cell epilogue (C tile layout: n = lane&15, m = r + 8*laneHi) ----
    const int n     = n0 + ln;
    const int mBase = m0 + wave * 16 + laneHi * 8;
    const float bi  = bsum[0 * H_ + n];
    const float bff = bsum[1 * H_ + n];
    const float bg  = bsum[2 * H_ + n];
    const float bor = bsum[3 * H_ + n];

    #pragma unroll
    for (int r = 0; r < 8; ++r) {
        int m = mBase + r;
        float gi = acc0[r] + bi;
        float gf = acc1[r] + bff;
        float gg = acc2[r] + bg;
        float go = acc3[r] + bor;
        float cold = cst[m * H_ + n];
        float cn = sigmf(gf) * cold + sigmf(gi) * tanhf(gg);
        float hn = sigmf(go) * tanhf(cn);
        cst[m * H_ + n] = cn;
        hst[m * H_ + n] = hn;
        unsigned short hb = f32_to_bf16(hn);
        hdst1[m * K2H + n] = hb;    // own [x|h] buffer, h half
        hdst2[m * K2H + n] = hb;    // next consumer's x half
    }
}

// ---------------------------------------------------------------------------
// out[b,t] = sigmoid(h1[b,:] . Wo + bo). One wave per batch row.
// ---------------------------------------------------------------------------
__global__ __launch_bounds__(256)
void out_proj(const float* __restrict__ h1, const float* __restrict__ Wo,
              const float* __restrict__ bo, float* __restrict__ outs, int t) {
    int lane = threadIdx.x & 31;
    int wave = threadIdx.x >> 5;
    int row  = blockIdx.x * 8 + wave;
    float s = 0.f;
    for (int j = lane; j < H_; j += 32) s += h1[row * H_ + j] * Wo[j];
    #pragma unroll
    for (int off = 16; off > 0; off >>= 1) s += __shfl_down(s, off, 32);
    if (lane == 0) outs[row * T_ + t] = sigmf(s + bo[0]);
}

// ---------------------------------------------------------------------------
extern "C" void kernel_launch(void* const* d_in, const int* in_sizes, int n_in,
                              void* d_out, int out_size, void* d_ws, size_t ws_size,
                              hipStream_t stream) {
    (void)in_sizes; (void)n_in; (void)out_size; (void)ws_size;
    const float* x0  = (const float*)d_in[0];   // decoder_input (B,1,H)
    const float* h0  = (const float*)d_in[1];   // (L,B,H)
    const float* c0  = (const float*)d_in[2];   // (L,B,H)
    const float* Wih = (const float*)d_in[3];   // (L,4H,H)
    const float* Whh = (const float*)d_in[4];   // (L,4H,H)
    const float* bih = (const float*)d_in[5];   // (L,4H)
    const float* bhh = (const float*)d_in[6];   // (L,4H)
    const float* Wo  = (const float*)d_in[7];   // (1,H)
    const float* bo  = (const float*)d_in[8];   // (1,)

    float* outs = (float*)d_out;                // (B,T)
    float* hst  = outs + B_ * T_;               // h_f region, used as live state
    float* cst  = hst + L_ * B_ * H_;           // c_f region, used as live state

    char* ws = (char*)d_ws;
    unsigned short* Wcat = (unsigned short*)ws;                               // L*4H*2H bf16
    float* bsum = (float*)(ws + (size_t)L_ * G4H * K2H * 2);                  // L*4H f32
    unsigned short* xh0 = (unsigned short*)((char*)bsum + (size_t)L_ * G4H * 4);
    unsigned short* xh1 = xh0 + (size_t)B_ * K2H;

    pack_weights<<<4096, 256, 0, stream>>>(Wih, Whh, Wcat);
    init_state<<<(L_ * B_ * H_ + 255) / 256, 256, 0, stream>>>(
        x0, h0, c0, bih, bhh, hst, cst, bsum, xh0, xh1);

    dim3 grid(H_ / 16, B_ / 128);
    for (int t = 0; t < T_; ++t) {
        // layer 0: reads xh0, writes h into xh0 h-half and xh1 x-half
        lstm_layer_step<<<grid, 256, 0, stream>>>(
            xh0, Wcat, bsum,
            hst, cst,
            xh0 + H_, xh1);
        // layer 1: reads xh1, writes h into xh1 h-half and xh0 x-half (next x)
        lstm_layer_step<<<grid, 256, 0, stream>>>(
            xh1, Wcat + (size_t)G4H * K2H, bsum + G4H,
            hst + B_ * H_, cst + B_ * H_,
            xh1 + H_, xh0);
        out_proj<<<B_ / 8, 256, 0, stream>>>(hst + B_ * H_, Wo, bo, outs, t);
    }
}
